// FCBoltzmannModel_37306085933570
// MI455X (gfx1250) — compile-verified, matrix operand order checked by reading, not run on hardware
//
#include <hip/hip_runtime.h>
#include <hip/hip_bf16.h>
#include <cstddef>
#include <cstdint>

// ---------------------------------------------------------------------------
// Annealed Gibbs sampling on a layered Boltzmann machine [vis, hid, out].
//   B=2048, IN=4096, HID=2048, OUT=512, STEPS=20
// Chain is independent per batch row: each 32-row batch tile runs all STEPS
// inside one workgroup with h/o state resident in LDS (~161 KB of the 320 KB
// CDNA5 WGP LDS). All GEMMs use v_wmma_f32_16x16x32_bf16 (wave32 WMMA).
// gemm0 (v @ W0^T) stages bf16 tiles into LDS with the Tensor Data Mover
// (tensor_load_to_lds + s_wait_tensorcnt, double-buffered) when available.
// ---------------------------------------------------------------------------

typedef __attribute__((ext_vector_type(16))) __bf16 v16bf;
typedef __attribute__((ext_vector_type(8)))  __bf16 v8bf;
typedef __attribute__((ext_vector_type(8)))  float  v8f;

constexpr int Bdim  = 2048;
constexpr int INdim = 4096;
constexpr int HID   = 2048;
constexpr int OUTd  = 512;
constexpr int OUTW  = INdim + HID + OUTd;  // 6656 packed output width

constexpr int GR    = 32;         // batch rows per Gibbs workgroup
constexpr int GWAVES= 16;         // 512-thread Gibbs blocks
constexpr int HSTR  = HID + 8;    // LDS row strides padded 16B to stagger banks
constexpr int OSTR  = OUTd + 8;

#if defined(__has_builtin)
#if __has_builtin(__builtin_amdgcn_tensor_load_to_lds) && \
    __has_builtin(__builtin_amdgcn_s_wait_tensorcnt)
#define HAVE_TDM 1
#endif
#endif

// Load 16 bf16 (two contiguous 16B chunks) into a WMMA fragment.
__device__ __forceinline__ v16bf load16bf(const __bf16* p0, const __bf16* p1) {
  v8bf a = *(const v8bf*)p0;
  v8bf b = *(const v8bf*)p1;
  v16bf r;
#pragma unroll
  for (int i = 0; i < 8; ++i) { r[i] = a[i]; r[i + 8] = b[i]; }
  return r;
}

__device__ __forceinline__ v8f wmma_bf16(v16bf a, v16bf b, v8f c) {
  return __builtin_amdgcn_wmma_f32_16x16x32_bf16(
      /*neg_a=*/false, a, /*neg_b=*/false, b,
      /*c_mod=*/(short)0, c, /*reuse_a=*/false, /*reuse_b=*/false);
}

#ifdef HAVE_TDM
typedef unsigned int u32x4 __attribute__((ext_vector_type(4)));
typedef int          i32x4 __attribute__((ext_vector_type(4)));
typedef int          i32x8 __attribute__((ext_vector_type(8)));

// TDM: DMA a 64x64 bf16 tile (row stride `strideElems`) from global memory to
// LDS at byte offset ldsOff. pad_enable appends 16B per 128B row, reproducing
// the 72-element LDS row stride used by the compute fragments.
// Tracked with TENSORcnt; EXEC ignored (issue from one wave only).
__device__ __forceinline__ void tdm_load_tile_64x64_bf16(
    const __bf16* gsrc, unsigned ldsOff, unsigned strideElems) {
  unsigned long long ga = (unsigned long long)(uintptr_t)gsrc;
  u32x4 g0;
  g0[0] = 1u;                                    // count=1 (valid descriptor)
  g0[1] = ldsOff;                                // lds_addr (bytes)
  g0[2] = (unsigned)(ga & 0xFFFFFFFFu);          // global_addr[31:0]
  g0[3] = (unsigned)((ga >> 32) & 0x1FFFFFFu)    // global_addr[56:32]
          | (2u << 30);                          // type = 2 ("image")
  i32x8 g1;
  g1[0] = (int)((1u << 16) |                     // data_size = 2 bytes
                (1u << 20) |                     // pad_enable
                (4u << 22) |                     // pad_interval = 32 DWORDs
                (3u << 25));                     // pad_amount   =  4 DWORDs
  g1[1] = (int)((64u & 0xFFFFu) << 16);          // tensor_dim0[15:0] = 64
  g1[2] = (int)((64u & 0xFFFFu) << 16);          // dim0[31:16]=0, tensor_dim1=64
  g1[3] = (int)(64u << 16);                      // dim1[31:16]=0, tile_dim0 = 64
  g1[4] = (int)(64u);                            // tile_dim1 = 64, tile_dim2 = 0
  g1[5] = (int)strideElems;                      // tensor_dim0_stride[31:0]
  g1[6] = 0;                                     // stride0[47:32], stride1[15:0]
  g1[7] = 0;                                     // stride1[47:16]
  i32x4 gz = {0, 0, 0, 0};
#if __clang_major__ >= 23
  i32x8 gz8 = {0, 0, 0, 0, 0, 0, 0, 0};
  __builtin_amdgcn_tensor_load_to_lds(g0, g1, gz, gz, gz8, 0);
#else
  __builtin_amdgcn_tensor_load_to_lds(g0, g1, gz, gz, 0);
#endif
}
#endif  // HAVE_TDM

// ---------------------------------------------------------------------------
// Kernel 1: W1 [OUT,HID] f32 -> bf16 in both orientations so both Gibbs GEMMs
// read contiguous per-lane K runs for the WMMA B fragment.
// ---------------------------------------------------------------------------
__global__ __launch_bounds__(256) void convert_w1_kernel(
    const float* __restrict__ W1, __bf16* __restrict__ W1b,
    __bf16* __restrict__ W1tb) {
  int i = blockIdx.x * 256 + threadIdx.x;
  if (i < OUTd * HID) {
    float w = W1[i];
    int r = i / HID, c = i % HID;
    W1b[i] = (__bf16)w;                      // [OUT,HID]  (gap_o B-frag rows)
    W1tb[(size_t)c * OUTd + r] = (__bf16)w;  // [HID,OUT]  (gap_h B-frag rows)
  }
}

// Elementwise f32 -> bf16 (for x and W0, feeding gemm0's TDM/LDS staging).
__global__ __launch_bounds__(256) void cvt_bf16_kernel(
    const float* __restrict__ in, __bf16* __restrict__ out, int n) {
  int i = blockIdx.x * 256 + threadIdx.x;
  if (i < n) out[i] = (__bf16)in[i];
}

// ---------------------------------------------------------------------------
// Kernel 2: vwb = x @ W0^T + b0. LDS-staged 64x64 block GEMM, K=4096.
// bf16 tiles DMA'd into LDS by the Tensor Data Mover (double-buffered,
// overlapped with WMMA compute); fragments come from ds_load_b128.
// 8 waves/block; wave w computes a 16x32 subtile (1 A-frag x 2 B-frags).
// ---------------------------------------------------------------------------
__global__ __launch_bounds__(256) void gemm0_kernel(
    const __bf16* __restrict__ xb, const __bf16* __restrict__ W0b,
    const float* __restrict__ b0, float* __restrict__ vwb) {
  constexpr int TSTR = 72;                    // 64 + 8 bf16 pad (16B) per row
  constexpr int TBUF = 64 * TSTR;             // elements per tile buffer

  const int tid  = threadIdx.x;
  const int lane = tid & 31;
  const int wave = tid >> 5;
  const int half = lane >> 4;                 // WMMA K-chunk selector
  const int nl   = lane & 15;
  const int m0 = blockIdx.y * 64;
  const int n0 = blockIdx.x * 64;
  const int msub  = wave >> 1;                // 0..3 : which 16-row A strip
  const int nbase = (wave & 1) * 32;          // 0/32 : which 32-col B strip

  v8f acc0 = {}, acc1 = {};

#ifdef HAVE_TDM
  __shared__ __bf16 As[2 * TBUF];             // double-buffered x tile
  __shared__ __bf16 Bs[2 * TBUF];             // double-buffered W0 tile
  const unsigned asOff = (unsigned)(uintptr_t)&As[0];  // low 32b = LDS offset
  const unsigned bsOff = (unsigned)(uintptr_t)&Bs[0];

  if (wave == 0) {  // prologue: fill buffer 0
    tdm_load_tile_64x64_bf16(xb  + (size_t)m0 * INdim, asOff, INdim);
    tdm_load_tile_64x64_bf16(W0b + (size_t)n0 * INdim, bsOff, INdim);
  }
  for (int kc = 0; kc < INdim; kc += 64) {
    const int cur = (kc >> 6) & 1;
    if (wave == 0) __builtin_amdgcn_s_wait_tensorcnt((short)0);
    __syncthreads();                          // buffer `cur` visible to all
    if (wave == 0 && kc + 64 < INdim) {       // DMA next tile during compute
      const int nxt = cur ^ 1;
      tdm_load_tile_64x64_bf16(xb  + (size_t)m0 * INdim + kc + 64,
                               asOff + (unsigned)(nxt * TBUF * 2), INdim);
      tdm_load_tile_64x64_bf16(W0b + (size_t)n0 * INdim + kc + 64,
                               bsOff + (unsigned)(nxt * TBUF * 2), INdim);
    }
    const __bf16* Ab = As + cur * TBUF;
    const __bf16* Bb = Bs + cur * TBUF;
#else
  __shared__ __bf16 As[TBUF];
  __shared__ __bf16 Bs[TBUF];
  const int sr = tid >> 2;                    // staging: row 0..63
  const int sc = (tid & 3) * 16;              // staging: 16-col chunk
  for (int kc = 0; kc < INdim; kc += 64) {
    __syncthreads();                          // previous fragment reads done
    *(v8bf*)&As[sr * TSTR + sc]     = *(const v8bf*)(xb  + (size_t)(m0 + sr) * INdim + kc + sc);
    *(v8bf*)&As[sr * TSTR + sc + 8] = *(const v8bf*)(xb  + (size_t)(m0 + sr) * INdim + kc + sc + 8);
    *(v8bf*)&Bs[sr * TSTR + sc]     = *(const v8bf*)(W0b + (size_t)(n0 + sr) * INdim + kc + sc);
    *(v8bf*)&Bs[sr * TSTR + sc + 8] = *(const v8bf*)(W0b + (size_t)(n0 + sr) * INdim + kc + sc + 8);
    __syncthreads();
    const __bf16* Ab = As;
    const __bf16* Bb = Bs;
#endif
#pragma unroll
    for (int kk = 0; kk < 64; kk += 32) {
      const int k0 = kk + half * 8, k1 = kk + 16 + half * 8;
      v16bf a   = load16bf(&Ab[(msub * 16 + nl) * TSTR + k0],
                           &Ab[(msub * 16 + nl) * TSTR + k1]);
      v16bf bf0 = load16bf(&Bb[(nbase + nl) * TSTR + k0],
                           &Bb[(nbase + nl) * TSTR + k1]);
      v16bf bf1 = load16bf(&Bb[(nbase + 16 + nl) * TSTR + k0],
                           &Bb[(nbase + 16 + nl) * TSTR + k1]);
      acc0 = wmma_bf16(a, bf0, acc0);
      acc1 = wmma_bf16(a, bf1, acc1);
    }
    __syncthreads();                          // reads done before refill
  }

  const int mg  = m0 + msub * 16 + half * 8;  // C/D layout: VGPR i -> M=i+8*half
  const int na  = n0 + nbase + nl;
  const int nb  = n0 + nbase + 16 + nl;
  const float ba = b0[na], bb = b0[nb];
#pragma unroll
  for (int i = 0; i < 8; ++i) {
    vwb[(size_t)(mg + i) * HID + na] = acc0[i] + ba;
    vwb[(size_t)(mg + i) * HID + nb] = acc1[i] + bb;
  }
}

// ---------------------------------------------------------------------------
// Kernel 3: persistent per-batch-tile annealed Gibbs chain, 32 rows/WG.
// ---------------------------------------------------------------------------
__global__ __launch_bounds__(512) void gibbs_kernel(
    const float* __restrict__ x, const float* __restrict__ b1,
    const float* __restrict__ u_h0, const float* __restrict__ u_o0,
    const float* __restrict__ u_h, const float* __restrict__ u_o,
    const __bf16* __restrict__ W1b, const __bf16* __restrict__ W1tb,
    const float* __restrict__ vwb, float* __restrict__ out, int steps) {
  __shared__ __bf16 h_s[GR * HSTR];   // 32 x 2056 bf16 = 128.5 KB
  __shared__ __bf16 o_s[GR * OSTR];   // 32 x  520 bf16 =  32.5 KB

  const int tid  = threadIdx.x;
  const int lane = tid & 31;
  const int wave = tid >> 5;           // 0..15
  const int half = lane >> 4;
  const int nl   = lane & 15;
  const int m0   = blockIdx.x * GR;
  const int mb   = half * 8;

  // Bernoulli(0.5) init of h and o.
  for (int i = tid; i < GR * HID; i += GWAVES * 32) {
    int r = i / HID, c = i % HID;
    h_s[r * HSTR + c] =
        (__bf16)((u_h0[(size_t)(m0 + r) * HID + c] < 0.5f) ? 1.0f : 0.0f);
  }
  for (int i = tid; i < GR * OUTd; i += GWAVES * 32) {
    int r = i / OUTd, c = i % OUTd;
    o_s[r * OSTR + c] =
        (__bf16)((u_o0[(size_t)(m0 + r) * OUTd + c] < 0.5f) ? 1.0f : 0.0f);
  }
  __syncthreads();

  const float g = 4.0f / (float)steps;   // (t_hi - t_lo)/steps
  for (int t = 0; t < steps; ++t) {
    const float T = g * (float)(steps - 1 - t) + 1.0f;
    const float invT = 1.0f / T;

    // ---- Phase 1: gap_h = o @ W1 + (vW0T + b0); sample h. N=HID, K=OUT ----
    constexpr int TILES_H = HID / (GWAVES * 16);  // 8 N-tiles per wave
    const __bf16* ap0 = o_s + nl * OSTR;          // A rows (M-tile 0)
    const __bf16* ap1 = o_s + (16 + nl) * OSTR;   // A rows (M-tile 1)
#pragma unroll 1
    for (int tt = 0; tt < TILES_H; ++tt) {
      const int ncol = (wave * TILES_H + tt) * 16 + nl;
      v8f acc0, acc1;
#pragma unroll
      for (int i = 0; i < 8; ++i) {
        acc0[i] = vwb[(size_t)(m0 + mb + i) * HID + ncol];
        acc1[i] = vwb[(size_t)(m0 + 16 + mb + i) * HID + ncol];
      }
      const __bf16* wrow = W1tb + (size_t)ncol * OUTd;
#pragma unroll 4
      for (int k = 0; k < OUTd; k += 32) {
        __builtin_prefetch(wrow + k + 64, 0, 3);
        const int k0 = k + half * 8, k1 = k + 16 + half * 8;
        v16bf b  = load16bf(wrow + k0, wrow + k1);   // shared by both M-tiles
        v16bf a0 = load16bf(ap0 + k0, ap0 + k1);
        v16bf a1 = load16bf(ap1 + k0, ap1 + k1);
        acc0 = wmma_bf16(a0, b, acc0);
        acc1 = wmma_bf16(a1, b, acc1);
      }
#pragma unroll
      for (int i = 0; i < 8; ++i) {
        const int ma = mb + i, mc = 16 + mb + i;
        const float sa = 1.0f / (1.0f + __expf(-acc0[i] * invT));
        const float sb = 1.0f / (1.0f + __expf(-acc1[i] * invT));
        const float ua = u_h[((size_t)t * Bdim + (m0 + ma)) * HID + ncol];
        const float ub = u_h[((size_t)t * Bdim + (m0 + mc)) * HID + ncol];
        h_s[ma * HSTR + ncol] = (__bf16)((ua < sa) ? 1.0f : 0.0f);
        h_s[mc * HSTR + ncol] = (__bf16)((ub < sb) ? 1.0f : 0.0f);
      }
    }
    __syncthreads();

    // ---- Phase 2: gap_o = h @ W1^T + b1; sample o. N=OUT, K=HID ----
    constexpr int TILES_O = OUTd / (GWAVES * 16);  // 2 N-tiles per wave
    const __bf16* bp0 = h_s + nl * HSTR;
    const __bf16* bp1 = h_s + (16 + nl) * HSTR;
#pragma unroll 1
    for (int tt = 0; tt < TILES_O; ++tt) {
      const int ncol = (wave * TILES_O + tt) * 16 + nl;
      const float bias = b1[ncol];
      v8f acc0, acc1;
#pragma unroll
      for (int i = 0; i < 8; ++i) { acc0[i] = bias; acc1[i] = bias; }
      const __bf16* wrow = W1b + (size_t)ncol * HID;
#pragma unroll 4
      for (int k = 0; k < HID; k += 32) {
        __builtin_prefetch(wrow + k + 64, 0, 3);
        const int k0 = k + half * 8, k1 = k + 16 + half * 8;
        v16bf b  = load16bf(wrow + k0, wrow + k1);
        v16bf a0 = load16bf(bp0 + k0, bp0 + k1);
        v16bf a1 = load16bf(bp1 + k0, bp1 + k1);
        acc0 = wmma_bf16(a0, b, acc0);
        acc1 = wmma_bf16(a1, b, acc1);
      }
#pragma unroll
      for (int i = 0; i < 8; ++i) {
        const int ma = mb + i, mc = 16 + mb + i;
        const float sa = 1.0f / (1.0f + __expf(-acc0[i] * invT));
        const float sb = 1.0f / (1.0f + __expf(-acc1[i] * invT));
        const float ua = u_o[((size_t)t * Bdim + (m0 + ma)) * OUTd + ncol];
        const float ub = u_o[((size_t)t * Bdim + (m0 + mc)) * OUTd + ncol];
        o_s[ma * OSTR + ncol] = (__bf16)((ua < sa) ? 1.0f : 0.0f);
        o_s[mc * OSTR + ncol] = (__bf16)((ub < sb) ? 1.0f : 0.0f);
      }
    }
    __syncthreads();
  }

  // Pack output: [v | h | o] per row, f32.
  for (int i = tid; i < GR * INdim; i += GWAVES * 32) {
    int r = i / INdim, c = i % INdim;
    out[(size_t)(m0 + r) * OUTW + c] = x[(size_t)(m0 + r) * INdim + c];
  }
  for (int i = tid; i < GR * HID; i += GWAVES * 32) {
    int r = i / HID, c = i % HID;
    out[(size_t)(m0 + r) * OUTW + INdim + c] = (float)h_s[r * HSTR + c];
  }
  for (int i = tid; i < GR * OUTd; i += GWAVES * 32) {
    int r = i / OUTd, c = i % OUTd;
    out[(size_t)(m0 + r) * OUTW + INdim + HID + c] = (float)o_s[r * OSTR + c];
  }
}

// ---------------------------------------------------------------------------
extern "C" void kernel_launch(void* const* d_in, const int* in_sizes, int n_in,
                              void* d_out, int out_size, void* d_ws, size_t ws_size,
                              hipStream_t stream) {
  const float* x    = (const float*)d_in[0];
  const float* W0   = (const float*)d_in[1];
  const float* b0   = (const float*)d_in[2];
  const float* W1   = (const float*)d_in[3];
  const float* b1   = (const float*)d_in[4];
  const float* u_h0 = (const float*)d_in[5];
  const float* u_o0 = (const float*)d_in[6];
  const float* u_h  = (const float*)d_in[7];
  const float* u_o  = (const float*)d_in[8];

  // STEPS derived from sizes (deterministic; steps scalar lives on device).
  const int steps = in_sizes[7] / in_sizes[5];  // (S*B*HID)/(B*HID)

  // Workspace partition.
  char* ws = (char*)d_ws;
  size_t off = 0;
  __bf16* W1b  = (__bf16*)(ws + off); off += (size_t)OUTd * HID * 2;   //  2 MB
  __bf16* W1tb = (__bf16*)(ws + off); off += (size_t)OUTd * HID * 2;   //  2 MB
  float*  vwb  = (float*) (ws + off); off += (size_t)Bdim * HID * 4;   // 16 MB
  __bf16* xb   = (__bf16*)(ws + off); off += (size_t)Bdim * INdim * 2; // 16 MB
  __bf16* W0b  = (__bf16*)(ws + off); off += (size_t)HID * INdim * 2;  // 16 MB

  float* out = (float*)d_out;

  convert_w1_kernel<<<(OUTd * HID + 255) / 256, 256, 0, stream>>>(W1, W1b, W1tb);
  cvt_bf16_kernel<<<(Bdim * INdim + 255) / 256, 256, 0, stream>>>(x, xb, Bdim * INdim);
  cvt_bf16_kernel<<<(HID * INdim + 255) / 256, 256, 0, stream>>>(W0, W0b, HID * INdim);

  dim3 g0(HID / 64, Bdim / 64);
  gemm0_kernel<<<g0, 256, 0, stream>>>(xb, W0b, b0, vwb);

  gibbs_kernel<<<Bdim / GR, 512, 0, stream>>>(x, b1, u_h0, u_o0, u_h, u_o,
                                              W1b, W1tb, vwb, out, steps);
}
